// AttLayer_21646635172403
// MI455X (gfx1250) — compile-verified, matrix operand order checked by reading, not run on hardware
//
#include <hip/hip_runtime.h>
#include <hip/hip_bf16.h>
#include <math.h>

// ---------------------------------------------------------------------------
// Multimodal transformer layer for MI455X (gfx1250, wave32, WMMA).
// All GEMMs use v_wmma_f32_16x16x32_bf16 (bf16 operands, f32 accumulate).
// Positional trick: scores = q.(k+pos_k) + pos_q.pos_k  -> fold pos into K,
// precompute PB[sq,sk] once per attention.
// ---------------------------------------------------------------------------

#define Bsz  16
#define Ssz  256
#define Hsz  768
#define NHsz 12
#define HDsz 64
#define Isz  3072

typedef __attribute__((ext_vector_type(16))) __bf16 bf16x16;
typedef __attribute__((ext_vector_type(8)))  __bf16 bf16x8;
typedef __attribute__((ext_vector_type(4)))  __bf16 bf16x4;
typedef __attribute__((ext_vector_type(8)))  float  floatx8;
typedef __attribute__((ext_vector_type(4)))  float  floatx4;

union Frag16 { bf16x16 v; bf16x8 h[2]; };

static __device__ inline floatx8 wmma_bf16(bf16x16 a, bf16x16 b, floatx8 c) {
  // (neg_a, A, neg_b, B, c_mod, C, reuse_a, reuse_b)
  return __builtin_amdgcn_wmma_f32_16x16x32_bf16(false, a, false, b,
                                                 (short)0, c, false, false);
}

// 4 floats -> 4 bf16 (lowers to v_cvt_pk_bf16_f32 pairs)
static __device__ inline bf16x4 pack4(floatx4 v) {
  bf16x4 r;
  r[0] = (__bf16)v[0]; r[1] = (__bf16)v[1];
  r[2] = (__bf16)v[2]; r[3] = (__bf16)v[3];
  return r;
}
// 8 consecutive f32 (16B-aligned) -> bf16x8 via two b128 loads + packed cvt
static __device__ inline bf16x8 load8_bf16(const float* p) {
  floatx4 a = *(const floatx4*)(p);
  floatx4 b = *(const floatx4*)(p + 4);
  union { bf16x8 v; bf16x4 q[2]; } u;
  u.q[0] = pack4(a); u.q[1] = pack4(b);
  return u.v;
}

// ---------------------------------------------------------------------------
// GEMM: C[M,N] = act(A[M,K] @ W[K,N] + bias[N]).  act: 0=none, 1=gelu(erf).
// Tile 64x64, 128 threads (4 waves), K-step 32, LDS staged bf16, WMMA.
// M%64==0, N%64==0, K%32==0 (all calls satisfy this).
// Wave w computes rows [w*16, w*16+16) x cols [0,64) of the tile.
// ---------------------------------------------------------------------------
__global__ __launch_bounds__(128) void gemm_bias_act_kernel(
    const float* __restrict__ A, const float* __restrict__ W,
    const float* __restrict__ bias, float* __restrict__ C,
    int M, int N, int K, int act)
{
  __shared__ __bf16 As[64][32];   // A tile, row-major
  __shared__ __bf16 Bt[64][32];   // W tile transposed: Bt[n][k]
  const int tid  = threadIdx.x;
  const int lane = tid & 31;
  const int wave = tid >> 5;
  const int tm0  = blockIdx.x * 64;
  const int tn0  = blockIdx.y * 64;
  const int lm   = lane & 15;     // m (or n) within 16-tile
  const int cb   = lane >> 4;     // K half selector (k-base 0 or 8)

  const floatx8 z = {0.f,0.f,0.f,0.f,0.f,0.f,0.f,0.f};
  floatx8 acc[4] = {z, z, z, z};

  for (int k0 = 0; k0 < K; k0 += 32) {
    // ---- batched loads into registers (coalesced, pipelined) ----
    floatx4 aT[4];
#pragma unroll
    for (int i = 0; i < 4; ++i) {
      int f  = tid + i * 128;            // float4 strip id, 0..511
      int r  = f >> 3;                   // row 0..63
      int c4 = (f & 7) << 2;             // col 0,4,..,28
      aT[i] = *(const floatx4*)(A + (size_t)(tm0 + r) * K + k0 + c4);
    }
    float bT[16];
#pragma unroll
    for (int i = 0; i < 4; ++i) {
      int s  = tid + i * 128;            // column-strip id, 0..511
      int n  = s & 63;                   // n within tile
      int kq = (s >> 6) << 2;            // k base 0,4,..,28
#pragma unroll
      for (int j = 0; j < 4; ++j)
        bT[i * 4 + j] = W[(size_t)(k0 + kq + j) * N + tn0 + n];
    }
    if (k0 + 32 < K) {                   // gfx1250 global_prefetch_b8
      __builtin_prefetch(A + (size_t)(tm0 + (tid >> 1)) * K + k0 + 32, 0, 1);
      __builtin_prefetch(W + (size_t)(k0 + 32 + (tid & 31)) * N + tn0, 0, 1);
    }
    // ---- convert + LDS stores (b64) ----
#pragma unroll
    for (int i = 0; i < 4; ++i) {
      int f  = tid + i * 128;
      int r  = f >> 3;
      int c4 = (f & 7) << 2;
      *(bf16x4*)(&As[r][c4]) = pack4(aT[i]);
    }
#pragma unroll
    for (int i = 0; i < 4; ++i) {
      int s  = tid + i * 128;
      int n  = s & 63;
      int kq = (s >> 6) << 2;
      floatx4 t = {bT[i*4+0], bT[i*4+1], bT[i*4+2], bT[i*4+3]};
      *(bf16x4*)(&Bt[n][kq]) = pack4(t);
    }
    __syncthreads();

    // A fragment: lane lm holds row (wave*16+lm); K runs [8*cb,8*cb+8) and +16
    Frag16 af;
    const bf16x8* arow = (const bf16x8*)(&As[wave * 16 + lm][0]);
    af.h[0] = arow[cb];
    af.h[1] = arow[2 + cb];
#pragma unroll
    for (int nb = 0; nb < 4; ++nb) {
      Frag16 bfv;
      const bf16x8* brow = (const bf16x8*)(&Bt[nb * 16 + lm][0]);
      bfv.h[0] = brow[cb];
      bfv.h[1] = brow[2 + cb];
      acc[nb] = wmma_bf16(af.v, bfv.v, acc[nb]);
    }
    __syncthreads();
  }

  // epilogue: C/D layout — VGPR r: m = r + 8*cb, n = lane&15 (+16*nb)
  const int mtop = cb * 8;
#pragma unroll
  for (int nb = 0; nb < 4; ++nb) {
    int gn = tn0 + nb * 16 + lm;
    float bv = bias[gn];
#pragma unroll
    for (int r = 0; r < 8; ++r) {
      int gm = tm0 + wave * 16 + mtop + r;
      float x = acc[nb][r] + bv;
      if (act) x = x * 0.5f * (1.0f + erff(x * 0.70710678f));
      C[(size_t)gm * N + gn] = x;
    }
  }
}

// ---------------------------------------------------------------------------
// Attention core: one wave per (b, h, 16-row q tile).
// scores = (Q @ K'^T)*scale + PB*scale + maskbias ; softmax ; @ V.
// K' already has pos folded in. Sk <= 512.  mask1 = [B,256]; mask2 covers
// sk >= 256 for the concat (Sk=512) case, else unused.
// ---------------------------------------------------------------------------
__global__ __launch_bounds__(32) void attn_kernel(
    const float* __restrict__ Q, const float* __restrict__ Kp,
    const float* __restrict__ V, const float* __restrict__ PB,
    const int* __restrict__ mask1, const int* __restrict__ mask2,
    float* __restrict__ O, int Sk, float scale)
{
  __shared__ float sc[16][512];
  __shared__ float rsum[16];
  const int bh = blockIdx.x;
  const int b  = bh / NHsz;
  const int h  = bh % NHsz;
  const int qt = blockIdx.y;             // q tile (16 rows)
  const int lane = threadIdx.x;
  const int lm = lane & 15;
  const int cb = lane >> 4;
  const int kb = cb * 8;

  const float* Qb = Q  + ((size_t)b * Ssz + qt * 16) * Hsz + h * HDsz;
  const float* Kb = Kp + (size_t)b * Sk * Hsz + h * HDsz;
  const float* Vb = V  + (size_t)b * Sk * Hsz + h * HDsz;

  // Q fragments for d=[0,32) and d=[32,64): vectorized f32->bf16 builds
  Frag16 qf[2];
#pragma unroll
  for (int hf = 0; hf < 2; ++hf) {
    const float* p = Qb + (size_t)lm * Hsz + hf * 32;
    qf[hf].h[0] = load8_bf16(p + kb);
    qf[hf].h[1] = load8_bf16(p + kb + 16);
  }

  const floatx8 z = {0.f,0.f,0.f,0.f,0.f,0.f,0.f,0.f};

  // --- scores ---
  for (int st = 0; st < Sk; st += 16) {
    const float* kp = Kb + (size_t)(st + lm) * Hsz;   // B = K'^T (n = sk)
    Frag16 kf0, kf1;
    kf0.h[0] = load8_bf16(kp + kb);
    kf0.h[1] = load8_bf16(kp + kb + 16);
    kf1.h[0] = load8_bf16(kp + 32 + kb);
    kf1.h[1] = load8_bf16(kp + 32 + kb + 16);
    floatx8 acc = z;
    acc = wmma_bf16(qf[0].v, kf0.v, acc);
    acc = wmma_bf16(qf[1].v, kf1.v, acc);
    int sk = st + lm;
    int mv = (sk < Ssz) ? mask1[b * Ssz + sk] : mask2[b * Ssz + (sk - Ssz)];
    float mb = mv ? 0.f : -1e30f;
#pragma unroll
    for (int r = 0; r < 8; ++r) {
      int qm = cb * 8 + r;
      sc[qm][sk] = acc[r] * scale + PB[(size_t)(qt * 16 + qm) * Sk + sk] * scale + mb;
    }
  }
  __syncthreads();

  // --- softmax over rows (wave32 shuffle reductions) ---
  for (int r = 0; r < 16; ++r) {
    float mx = -3.0e38f;
    for (int i = lane; i < Sk; i += 32) mx = fmaxf(mx, sc[r][i]);
#pragma unroll
    for (int o = 16; o > 0; o >>= 1) mx = fmaxf(mx, __shfl_xor(mx, o, 32));
    float s = 0.f;
    for (int i = lane; i < Sk; i += 32) {
      float e = __expf(sc[r][i] - mx);
      sc[r][i] = e; s += e;
    }
#pragma unroll
    for (int o = 16; o > 0; o >>= 1) s += __shfl_xor(s, o, 32);
    if (lane == 0) rsum[r] = s;
  }
  __syncthreads();

  // --- P @ V ---
  floatx8 oacc[4] = {z, z, z, z};
  for (int st = 0; st < Sk; st += 32) {
    Frag16 pf;                           // A = probs [16 x 32], from LDS
    pf.h[0] = load8_bf16(&sc[lm][st + kb]);
    pf.h[1] = load8_bf16(&sc[lm][st + kb + 16]);
#pragma unroll
    for (int nb = 0; nb < 4; ++nb) {
      float vt[16];                      // batched column loads (coalesced)
#pragma unroll
      for (int j = 0; j < 16; ++j) {
        int kk = kb + (j & 7) + ((j >> 3) << 4);
        vt[j] = Vb[(size_t)(st + kk) * Hsz + nb * 16 + lm];
      }
      bf16x16 vf;
#pragma unroll
      for (int j = 0; j < 16; ++j) vf[j] = (__bf16)vt[j];
      oacc[nb] = wmma_bf16(pf.v, vf, oacc[nb]);
    }
  }
#pragma unroll
  for (int nb = 0; nb < 4; ++nb)
#pragma unroll
    for (int r = 0; r < 8; ++r) {
      int qm = cb * 8 + r;
      float inv = 1.0f / rsum[qm];
      O[((size_t)b * Ssz + qt * 16 + qm) * Hsz + h * HDsz + nb * 16 + lm] =
          oacc[nb][r] * inv;
    }
}

// ---------------------------------------------------------------------------
// y = LayerNorm(x + res) * g + beta ; one 256-thread block per 768-wide row.
// ---------------------------------------------------------------------------
__global__ __launch_bounds__(256) void add_ln_kernel(
    const float* __restrict__ X, const float* __restrict__ R,
    const float* __restrict__ g, const float* __restrict__ beta,
    float* __restrict__ Y)
{
  __shared__ float red[8];
  __shared__ float stat[2];
  const int row = blockIdx.x;
  const int tid = threadIdx.x;
  const float* x  = X + (size_t)row * Hsz;
  const float* rr = R + (size_t)row * Hsz;
  float vals[3];
  float s = 0.f;
#pragma unroll
  for (int i = 0; i < 3; ++i) {
    int c = tid + i * 256;
    float t = x[c] + rr[c];
    vals[i] = t; s += t;
  }
#pragma unroll
  for (int o = 16; o > 0; o >>= 1) s += __shfl_xor(s, o, 32);
  if ((tid & 31) == 0) red[tid >> 5] = s;
  __syncthreads();
  if (tid == 0) { float t = 0.f; for (int i = 0; i < 8; ++i) t += red[i];
                  stat[0] = t * (1.0f / Hsz); }
  __syncthreads();
  float mean = stat[0];
  float vs = 0.f;
#pragma unroll
  for (int i = 0; i < 3; ++i) { float d = vals[i] - mean; vs += d * d; }
#pragma unroll
  for (int o = 16; o > 0; o >>= 1) vs += __shfl_xor(vs, o, 32);
  if ((tid & 31) == 0) red[tid >> 5] = vs;
  __syncthreads();
  if (tid == 0) { float t = 0.f; for (int i = 0; i < 8; ++i) t += red[i];
                  stat[1] = rsqrtf(t * (1.0f / Hsz) + 1e-12f); }
  __syncthreads();
  float inv = stat[1];
#pragma unroll
  for (int i = 0; i < 3; ++i) {
    int c = tid + i * 256;
    Y[(size_t)row * Hsz + c] = (vals[i] - mean) * inv * g[c] + beta[c];
  }
}

// PB[sq,sk] = dot(pos[sq], pos[sk]) over HD=64
__global__ __launch_bounds__(256) void posbias_kernel(
    const float* __restrict__ pos, float* __restrict__ PB, int Sk)
{
  int idx = blockIdx.x * 256 + threadIdx.x;
  if (idx >= Ssz * Sk) return;
  int sq = idx / Sk, sk = idx - sq * Sk;
  const float* a = pos + sq * HDsz;
  const float* b = pos + sk * HDsz;
  float s = 0.f;
#pragma unroll
  for (int d = 0; d < HDsz; ++d) s += a[d] * b[d];
  PB[idx] = s;
}

// K[b,s,h*64+d] += pos[s,d]  (pos broadcast across batch and heads)
__global__ __launch_bounds__(256) void addpos_kernel(
    float* __restrict__ Kt, const float* __restrict__ pos, int Sk, int total)
{
  int idx = blockIdx.x * 256 + threadIdx.x;
  if (idx >= total) return;
  int c = idx % Hsz;
  int s = (idx / Hsz) % Sk;
  Kt[idx] += pos[s * HDsz + (c & 63)];
}

// C[b, 0..2S, H] = concat(A[b,:,:], Bv[b,:,:]) along seq
__global__ __launch_bounds__(256) void concat_kernel(
    const float* __restrict__ A, const float* __restrict__ Bv,
    float* __restrict__ C)
{
  int idx = blockIdx.x * 256 + threadIdx.x;
  const int total = Bsz * 2 * Ssz * Hsz;
  if (idx >= total) return;
  int c  = idx % Hsz;
  int bs = idx / Hsz;
  int s  = bs % (2 * Ssz);
  int b  = bs / (2 * Ssz);
  C[idx] = (s < Ssz) ? A[((size_t)b * Ssz + s) * Hsz + c]
                     : Bv[((size_t)b * Ssz + (s - Ssz)) * Hsz + c];
}

// ---------------------------------------------------------------------------
// Host orchestration
// ---------------------------------------------------------------------------
struct AttW { const float *qw,*qb,*kw,*kb,*vw,*vb,*pos; };
struct LNW  { const float *w,*b,*g,*beta; };
struct LinW { const float *w,*b; };
struct Mod  { AttW satt, catt; LNW sout, cout_; LinW uinter, inter; LNW uout, out; };

static inline const float* F(void* const* d, int i) { return (const float*)d[i]; }

// Params pytree assumed flattened in JAX sorted-key order:
// modalities: audio(9), text(43), vision(77); per modality (34 leaves):
// cross.att{k_b,k_w,pos,q_b,q_w,v_b,v_w} cross.out{b,beta,g,w}
// inter{b,w} out{b,beta,g,w} self.att{...} self.out{...} uni_inter{b,w} uni_out{...}
static Mod load_mod(void* const* d, int P) {
  Mod m;
  m.catt   = { F(d,P+4),  F(d,P+3),  F(d,P+1),  F(d,P+0),
               F(d,P+6),  F(d,P+5),  F(d,P+2) };
  m.cout_  = { F(d,P+10), F(d,P+7),  F(d,P+9),  F(d,P+8) };
  m.inter  = { F(d,P+12), F(d,P+11) };
  m.out    = { F(d,P+16), F(d,P+13), F(d,P+15), F(d,P+14) };
  m.satt   = { F(d,P+21), F(d,P+20), F(d,P+18), F(d,P+17),
               F(d,P+23), F(d,P+22), F(d,P+19) };
  m.sout   = { F(d,P+27), F(d,P+24), F(d,P+26), F(d,P+25) };
  m.uinter = { F(d,P+29), F(d,P+28) };
  m.uout   = { F(d,P+33), F(d,P+30), F(d,P+32), F(d,P+31) };
  return m;
}

extern "C" void kernel_launch(void* const* d_in, const int* in_sizes, int n_in,
                              void* d_out, int out_size, void* d_ws, size_t ws_size,
                              hipStream_t stream) {
  (void)in_sizes; (void)n_in; (void)out_size; (void)ws_size;
  const int BS = Bsz * Ssz;              // 4096 rows
  const float scale = 0.125f;            // 1/sqrt(64)

  const float* ti  = F(d_in, 0);
  const float* tui = F(d_in, 1);
  const int*   tm  = (const int*)d_in[2];
  const float* ai  = F(d_in, 3);
  const float* aui = F(d_in, 4);
  const int*   am  = (const int*)d_in[5];
  const float* vi  = F(d_in, 6);
  const float* vui = F(d_in, 7);
  const int*   vm  = (const int*)d_in[8];
  Mod Pa = load_mod(d_in, 9);
  Mod Pt = load_mod(d_in, 43);
  Mod Pv = load_mod(d_in, 77);

  float* out = (float*)d_out;
  const size_t OSZ = (size_t)BS * Hsz;   // 3,145,728 elems per output tensor

  // workspace bump allocator (~215 MB total)
  char* wsp = (char*)d_ws;
  size_t off = 0;
  auto alloc = [&](size_t n) {
    float* p = (float*)(wsp + off);
    off = (off + n * sizeof(float) + 255) & ~(size_t)255;
    return p;
  };
  float* t_sa = alloc(OSZ);
  float* a_sa = alloc(OSZ);
  float* v_sa = alloc(OSZ);
  float* qb_  = alloc(OSZ);
  float* kb_  = alloc(2 * OSZ);          // up to Sk=512
  float* vb_  = alloc(2 * OSZ);
  float* ao_  = alloc(OSZ);              // attention output
  float* pj_  = alloc(OSZ);              // projection scratch
  float* cx_  = alloc(2 * OSZ);          // concat ctx
  float* hid  = alloc((size_t)BS * Isz); // FFN hidden
  float* pb_  = alloc((size_t)Ssz * 2 * Ssz); // pos bias (256x512 max)
  float* su_  = alloc(OSZ);              // uni-SA / cross-att result

  auto gemm = [&](const float* A, const float* W, const float* b, float* C,
                  int M, int N, int K, int act) {
    gemm_bias_act_kernel<<<dim3(M / 64, N / 64), dim3(128), 0, stream>>>(
        A, W, b, C, M, N, K, act);
  };
  auto attblock = [&](const float* x, const float* ctx, int Sk,
                      const int* m1, const int* m2,
                      const AttW& ap, const LNW& op, float* dest) {
    gemm(x,   ap.qw, ap.qb, qb_, BS, Hsz, Hsz, 0);
    gemm(ctx, ap.kw, ap.kb, kb_, Bsz * Sk, Hsz, Hsz, 0);
    int ktot = Bsz * Sk * Hsz;
    addpos_kernel<<<(ktot + 255) / 256, 256, 0, stream>>>(kb_, ap.pos, Sk, ktot);
    gemm(ctx, ap.vw, ap.vb, vb_, Bsz * Sk, Hsz, Hsz, 0);
    int ptot = Ssz * Sk;
    posbias_kernel<<<(ptot + 255) / 256, 256, 0, stream>>>(ap.pos, pb_, Sk);
    attn_kernel<<<dim3(Bsz * NHsz, Ssz / 16), dim3(32), 0, stream>>>(
        qb_, kb_, vb_, pb_, m1, m2, ao_, Sk, scale);
    gemm(ao_, op.w, op.b, pj_, BS, Hsz, Hsz, 0);
    add_ln_kernel<<<BS, 256, 0, stream>>>(pj_, x, op.g, op.beta, dest);
  };
  auto ffn = [&](const float* x, const LinW& pi, const LNW& po, float* dest) {
    gemm(x, pi.w, pi.b, hid, BS, Isz, Hsz, 1);   // gelu
    gemm(hid, po.w, po.b, pj_, BS, Hsz, Isz, 0);
    add_ln_kernel<<<BS, 256, 0, stream>>>(pj_, x, po.g, po.beta, dest);
  };

  // ---- self-attention + unimodal FFN, per modality ----
  attblock(ti,  ti,  Ssz, tm, nullptr, Pt.satt, Pt.sout, t_sa);
  attblock(tui, tui, Ssz, tm, nullptr, Pt.satt, Pt.sout, su_);
  ffn(su_, Pt.uinter, Pt.uout, out + 3 * OSZ);        // tu_out

  attblock(ai,  ai,  Ssz, am, nullptr, Pa.satt, Pa.sout, a_sa);
  attblock(aui, aui, Ssz, am, nullptr, Pa.satt, Pa.sout, su_);
  ffn(su_, Pa.uinter, Pa.uout, out + 4 * OSZ);        // au_out

  attblock(vi,  vi,  Ssz, vm, nullptr, Pv.satt, Pv.sout, v_sa);
  attblock(vui, vui, Ssz, vm, nullptr, Pv.satt, Pv.sout, su_);
  ffn(su_, Pv.uinter, Pv.uout, out + 5 * OSZ);        // vu_out

  // ---- cross attention + FFN ----
  {
    const int ctot = Bsz * 2 * Ssz * Hsz;
    concat_kernel<<<(ctot + 255) / 256, 256, 0, stream>>>(a_sa, v_sa, cx_);
  }
  attblock(t_sa, cx_,  2 * Ssz, am, vm,      Pt.catt, Pt.cout_, su_);
  ffn(su_, Pt.inter, Pt.out, out + 0 * OSZ);          // t_out

  attblock(a_sa, t_sa, Ssz,     tm, nullptr, Pa.catt, Pa.cout_, su_);
  ffn(su_, Pa.inter, Pa.out, out + 1 * OSZ);          // a_out

  attblock(v_sa, t_sa, Ssz,     tm, nullptr, Pv.catt, Pv.cout_, su_);
  ffn(su_, Pv.inter, Pv.out, out + 2 * OSZ);          // v_out
}
